// MS_SSA_Conv_7988639171016
// MI455X (gfx1250) — matrix operand order repaired and checked by reading, not compiled, not run on hardware
//
#include <hip/hip_runtime.h>

// Problem constants (from reference)
#define T_  4
#define C_  512
#define Nn  196          // H*W tokens
#define NP  208          // Nn padded to multiple of 16 (13 tiles)

typedef _Float16 v16h __attribute__((ext_vector_type(16)));
typedef _Float16 v8h  __attribute__((ext_vector_type(8)));
typedef float    v8f  __attribute__((ext_vector_type(8)));

// ---------------------------------------------------------------------------
// Weight f32 -> f16 conversion (one 512x512 matrix per launch)
// ---------------------------------------------------------------------------
__global__ void wconv_kernel(const float* __restrict__ src, _Float16* __restrict__ dst) {
    int i = blockIdx.x * 256 + threadIdx.x;           // 262144 elements exactly
    dst[i] = (_Float16)src[i];
}

// ---------------------------------------------------------------------------
// BN scale/shift precompute for the 4 BN layers; proj bias folded into shift.
// scale = gamma / sqrt(var + eps); shift = beta - mean*scale  (+ b*scale for proj)
// ---------------------------------------------------------------------------
__global__ void bn_prep_kernel(
    const float* qg, const float* qb, const float* qm, const float* qv,
    const float* kg, const float* kb, const float* km, const float* kv,
    const float* vg, const float* vb, const float* vm, const float* vv,
    const float* pg, const float* pb, const float* pm, const float* pv,
    const float* proj_b, float* __restrict__ scale_all, float* __restrict__ shift_all) {
    int c = blockIdx.x * 256 + threadIdx.x;
    if (c >= C_) return;
    const float* G[4] = {qg, kg, vg, pg};
    const float* B[4] = {qb, kb, vb, pb};
    const float* M[4] = {qm, km, vm, pm};
    const float* V[4] = {qv, kv, vv, pv};
    for (int j = 0; j < 4; ++j) {
        float sc = G[j][c] * __frsqrt_rn(V[j][c] + 1e-5f);
        float sh = B[j][c] - M[j][c] * sc;
        if (j == 3) sh += proj_b[c] * sc;   // conv bias goes through BN scale
        scale_all[j * C_ + c] = sc;
        shift_all[j * C_ + c] = sh;
    }
}

// ---------------------------------------------------------------------------
// LIF over T on input x [T,C,N] -> spikes f16, stored transposed [t][n][c]
// (n padded to NP with zeros so WMMA B-tiles can read pad columns safely)
// ---------------------------------------------------------------------------
__global__ void lif_x_kernel(const float* __restrict__ x, _Float16* __restrict__ xs16) {
    int i = blockIdx.x * 256 + threadIdx.x;   // NP*C_ = 106496 -> 416 blocks
    int c = i & (C_ - 1);
    int n = i >> 9;
    if (n >= Nn) {
        for (int t = 0; t < T_; ++t) xs16[(size_t)t * NP * C_ + n * C_ + c] = (_Float16)0.0f;
        return;
    }
    float v = 0.0f;
    for (int t = 0; t < T_; ++t) {
        float xt = x[((size_t)t * C_ + c) * Nn + n];
        v += (xt - v) * 0.5f;                 // tau = 2
        float s = (v >= 1.0f) ? 1.0f : 0.0f;  // v_th = 1
        v = (s > 0.0f) ? 0.0f : v;            // hard reset
        xs16[(size_t)t * NP * C_ + n * C_ + c] = (_Float16)s;
    }
}

// ---------------------------------------------------------------------------
// q/k/v GEMM: U[o,n] = sum_c W[o,c] * xs[c,n], BN epilogue.
// Per-wave 16x16 tile via v_wmma_f32_16x16x32_f16, K = 512 in 16 steps.
// blockIdx.z = t*3 + widx (widx: 0=q,1=k,2=v)
// ---------------------------------------------------------------------------
__global__ __launch_bounds__(256)
void gemm_qkv_kernel(const _Float16* __restrict__ w16,
                     const _Float16* __restrict__ xs16,
                     const float* __restrict__ scale_all,
                     const float* __restrict__ shift_all,
                     float* __restrict__ u) {
    const int lane = threadIdx.x & 31;
    const int wave = threadIdx.x >> 5;
    const int widx = blockIdx.z % 3;
    const int t    = blockIdx.z / 3;
    const int mbase = blockIdx.x * 128 + wave * 16;
    const int mrow  = mbase + (lane & 15);
    const int kgrp  = lane >> 4;                 // half-wave K selector
    const int ncol  = blockIdx.y * 16 + (lane & 15);

    // A (weights, row-major [o,c]): lane 0-15 -> K {0..7,16..23}, lane 16-31 -> K {8..15,24..31}
    const _Float16* A = w16 + (size_t)widx * C_ * C_ + (size_t)mrow * C_ + kgrp * 8;
    // B (xs transposed [n][c]): lane 0-15 -> K {0..7,16..23}, lane 16-31 -> K {8..15,24..31}
    const _Float16* B = xs16 + (size_t)t * NP * C_ + (size_t)ncol * C_ + kgrp * 8;

    v8f acc = {};
#pragma unroll
    for (int kb = 0; kb < C_; kb += 32) {
        __builtin_prefetch((const void*)(A + kb + 64), 0, 0);
        __builtin_prefetch((const void*)(B + kb + 64), 0, 0);
        v8h a0 = *(const v8h*)(A + kb);
        v8h a1 = *(const v8h*)(A + kb + 16);
        v8h b0 = *(const v8h*)(B + kb);
        v8h b1 = *(const v8h*)(B + kb + 16);
        v16h Av, Bv;
#pragma unroll
        for (int i = 0; i < 8; ++i) { Av[i] = a0[i]; Av[i + 8] = a1[i];
                                      Bv[i] = b0[i]; Bv[i + 8] = b1[i]; }
        acc = __builtin_amdgcn_wmma_f32_16x16x32_f16(false, Av, false, Bv,
                                                     (short)0, acc, false, false);
    }

    if (ncol < Nn) {
#pragma unroll
        for (int r = 0; r < 8; ++r) {
            int o = mbase + r + (lane >> 4) * 8;   // D layout: M = r + 8*(lane>=16)
            float res = acc[r] * scale_all[widx * C_ + o] + shift_all[widx * C_ + o];
            u[(((size_t)widx * T_ + t) * C_ + o) * Nn + ncol] = res;
        }
    }
}

// ---------------------------------------------------------------------------
// LIF over T for q/k/v membrane inputs. q spikes -> f16 transposed [t][n][c];
// k,v spikes -> product k_s*v_s stored [t][c][n] for the kv reduction.
// ---------------------------------------------------------------------------
__global__ void lif_qkv_kernel(const float* __restrict__ u,
                               _Float16* __restrict__ qs16,
                               float* __restrict__ prod) {
    int i = blockIdx.x * 256 + threadIdx.x;   // over NP*C_
    int c = i & (C_ - 1);
    int n = i >> 9;
    if (n >= Nn) {
        for (int t = 0; t < T_; ++t) qs16[(size_t)t * NP * C_ + n * C_ + c] = (_Float16)0.0f;
        return;
    }
    float vq = 0.0f, vk = 0.0f, vv = 0.0f;
    for (int t = 0; t < T_; ++t) {
        size_t base = ((size_t)t * C_ + c) * Nn + n;
        float uq = u[base];
        float uk = u[(size_t)T_ * C_ * Nn + base];
        float uv = u[2 * (size_t)T_ * C_ * Nn + base];
        vq += (uq - vq) * 0.5f;  float sq = (vq >= 1.0f) ? 1.0f : 0.0f;  vq = (sq > 0.0f) ? 0.0f : vq;
        vk += (uk - vk) * 0.5f;  float sk = (vk >= 1.0f) ? 1.0f : 0.0f;  vk = (sk > 0.0f) ? 0.0f : vk;
        vv += (uv - vv) * 0.5f;  float sv = (vv >= 1.0f) ? 1.0f : 0.0f;  vv = (sv > 0.0f) ? 0.0f : vv;
        qs16[(size_t)t * NP * C_ + n * C_ + c] = (_Float16)sq;
        prod[base] = sk * sv;
    }
}

// ---------------------------------------------------------------------------
// kv[t,c] = sum_n k_s*v_s ; then LIF over T with v_th=0.5 -> kvs[t,c] in {0,1}
// One block (256 threads) per channel c.
// ---------------------------------------------------------------------------
__global__ void kv_reduce_kernel(const float* __restrict__ prod, float* __restrict__ kvs) {
    __shared__ float red[256];
    const int c = blockIdx.x;
    const int tid = threadIdx.x;
    float kvt[T_];
    for (int t = 0; t < T_; ++t) {
        red[tid] = (tid < Nn) ? prod[((size_t)t * C_ + c) * Nn + tid] : 0.0f;
        __syncthreads();
        for (int off = 128; off > 0; off >>= 1) {
            if (tid < off) red[tid] += red[tid + off];
            __syncthreads();
        }
        kvt[t] = red[0];
        __syncthreads();
    }
    if (tid == 0) {
        float v = 0.0f;
        for (int t = 0; t < T_; ++t) {
            v += (kvt[t] - v) * 0.5f;
            float s = (v >= 0.5f) ? 1.0f : 0.0f;   // talking_heads_lif v_th = 0.5
            v = (s > 0.0f) ? 0.0f : v;
            kvs[t * C_ + c] = s;
        }
    }
}

// ---------------------------------------------------------------------------
// y[t,n,c] = q_s[t,n,c] * kvs[t,c]  (binary mask), f16 transposed layout
// ---------------------------------------------------------------------------
__global__ void form_y_kernel(const _Float16* __restrict__ qs16,
                              const float* __restrict__ kvs,
                              _Float16* __restrict__ y16) {
    int i = blockIdx.x * 256 + threadIdx.x;   // T_*NP*C_ = 425984 -> 1664 blocks
    int c = i & (C_ - 1);
    int t = i / (NP * C_);
    y16[i] = (kvs[t * C_ + c] >= 0.5f) ? qs16[i] : (_Float16)0.0f;
}

// ---------------------------------------------------------------------------
// proj GEMM + bias/BN (prefused) + identity residual -> d_out [T,C,H,W]
// ---------------------------------------------------------------------------
__global__ __launch_bounds__(256)
void gemm_proj_kernel(const _Float16* __restrict__ w16p,
                      const _Float16* __restrict__ y16,
                      const float* __restrict__ scale_p,
                      const float* __restrict__ shift_p,
                      const float* __restrict__ x,
                      float* __restrict__ out) {
    const int lane = threadIdx.x & 31;
    const int wave = threadIdx.x >> 5;
    const int t    = blockIdx.z;
    const int mbase = blockIdx.x * 128 + wave * 16;
    const int mrow  = mbase + (lane & 15);
    const int kgrp  = lane >> 4;
    const int ncol  = blockIdx.y * 16 + (lane & 15);

    const _Float16* A = w16p + (size_t)mrow * C_ + kgrp * 8;
    const _Float16* B = y16 + (size_t)t * NP * C_ + (size_t)ncol * C_ + kgrp * 8;

    v8f acc = {};
#pragma unroll
    for (int kb = 0; kb < C_; kb += 32) {
        __builtin_prefetch((const void*)(A + kb + 64), 0, 0);
        __builtin_prefetch((const void*)(B + kb + 64), 0, 0);
        v8h a0 = *(const v8h*)(A + kb);
        v8h a1 = *(const v8h*)(A + kb + 16);
        v8h b0 = *(const v8h*)(B + kb);
        v8h b1 = *(const v8h*)(B + kb + 16);
        v16h Av, Bv;
#pragma unroll
        for (int i = 0; i < 8; ++i) { Av[i] = a0[i]; Av[i + 8] = a1[i];
                                      Bv[i] = b0[i]; Bv[i + 8] = b1[i]; }
        acc = __builtin_amdgcn_wmma_f32_16x16x32_f16(false, Av, false, Bv,
                                                     (short)0, acc, false, false);
    }

    if (ncol < Nn) {
#pragma unroll
        for (int r = 0; r < 8; ++r) {
            int o = mbase + r + (lane >> 4) * 8;
            size_t oi = ((size_t)t * C_ + o) * Nn + ncol;
            out[oi] = acc[r] * scale_p[o] + shift_p[o] + x[oi];
        }
    }
}

// ---------------------------------------------------------------------------
extern "C" void kernel_launch(void* const* d_in, const int* in_sizes, int n_in,
                              void* d_out, int out_size, void* d_ws, size_t ws_size,
                              hipStream_t stream) {
    const float* x      = (const float*)d_in[0];
    const float* q_w    = (const float*)d_in[1];
    const float* k_w    = (const float*)d_in[2];
    const float* v_w    = (const float*)d_in[3];
    const float* proj_w = (const float*)d_in[4];
    const float* proj_b = (const float*)d_in[5];
    const float* qg = (const float*)d_in[6],  *qb = (const float*)d_in[7];
    const float* qm = (const float*)d_in[8],  *qv = (const float*)d_in[9];
    const float* kg = (const float*)d_in[10], *kb = (const float*)d_in[11];
    const float* km = (const float*)d_in[12], *kv = (const float*)d_in[13];
    const float* vg = (const float*)d_in[14], *vb = (const float*)d_in[15];
    const float* vm = (const float*)d_in[16], *vv = (const float*)d_in[17];
    const float* pg = (const float*)d_in[18], *pb = (const float*)d_in[19];
    const float* pm = (const float*)d_in[20], *pv = (const float*)d_in[21];

    char* ws = (char*)d_ws;
    size_t off = 0;
    _Float16* w16 = (_Float16*)(ws + off); off += (size_t)4 * C_ * C_ * sizeof(_Float16);      // 2 MB
    float* scale_all = (float*)(ws + off); off += (size_t)4 * C_ * sizeof(float);
    float* shift_all = (float*)(ws + off); off += (size_t)4 * C_ * sizeof(float);
    _Float16* xs16 = (_Float16*)(ws + off); off += (size_t)T_ * NP * C_ * sizeof(_Float16);
    float* u = (float*)(ws + off);          off += (size_t)3 * T_ * C_ * Nn * sizeof(float);
    _Float16* qs16 = (_Float16*)(ws + off); off += (size_t)T_ * NP * C_ * sizeof(_Float16);
    float* prod = (float*)(ws + off);       off += (size_t)T_ * C_ * Nn * sizeof(float);
    float* kvs = (float*)(ws + off);        off += (size_t)T_ * C_ * sizeof(float);
    _Float16* y16 = (_Float16*)(ws + off);  off += (size_t)T_ * NP * C_ * sizeof(_Float16);

    // 1) weights -> f16 (q,k,v,proj packed consecutively)
    wconv_kernel<<<1024, 256, 0, stream>>>(q_w,    w16 + 0 * (size_t)C_ * C_);
    wconv_kernel<<<1024, 256, 0, stream>>>(k_w,    w16 + 1 * (size_t)C_ * C_);
    wconv_kernel<<<1024, 256, 0, stream>>>(v_w,    w16 + 2 * (size_t)C_ * C_);
    wconv_kernel<<<1024, 256, 0, stream>>>(proj_w, w16 + 3 * (size_t)C_ * C_);

    // 2) BN scale/shift
    bn_prep_kernel<<<2, 256, 0, stream>>>(qg, qb, qm, qv, kg, kb, km, kv,
                                          vg, vb, vm, vv, pg, pb, pm, pv,
                                          proj_b, scale_all, shift_all);

    // 3) shortcut LIF -> spike tensor (f16, transposed, padded)
    lif_x_kernel<<<(NP * C_) / 256, 256, 0, stream>>>(x, xs16);

    // 4) q/k/v GEMMs + BN (WMMA)
    gemm_qkv_kernel<<<dim3(4, NP / 16, T_ * 3), 256, 0, stream>>>(w16, xs16, scale_all, shift_all, u);

    // 5) LIF over T for q/k/v
    lif_qkv_kernel<<<(NP * C_) / 256, 256, 0, stream>>>(u, qs16, prod);

    // 6) per-channel kv reduction + talking-heads LIF
    kv_reduce_kernel<<<C_, 256, 0, stream>>>(prod, kvs);

    // 7) y = q_s masked by kvs
    form_y_kernel<<<(T_ * NP * C_) / 256, 256, 0, stream>>>(qs16, kvs, y16);

    // 8) proj GEMM + bias/BN + residual (WMMA)
    gemm_proj_kernel<<<dim3(4, NP / 16, T_), 256, 0, stream>>>(
        w16 + 3 * (size_t)C_ * C_, y16, scale_all + 3 * C_, shift_all + 3 * C_, x, (float*)d_out);
}
